// DWL_61924838474335
// MI455X (gfx1250) — matrix operand order repaired and checked by reading, not compiled, not run on hardware
//
#include <hip/hip_runtime.h>
#include <cstddef>
#include <cstdint>

#define OBSD 128
#define PROJ 64
#define HID  256
#define LAT  24
#define G3   768
#define BB   1024
#define TT   256

// ---------- CDNA5 WMMA / TDM types ----------
typedef __attribute__((ext_vector_type(16))) __bf16    v16bf;
typedef __attribute__((ext_vector_type(8)))  float     v8f;
typedef __attribute__((ext_vector_type(4)))  unsigned  uv4;
typedef __attribute__((ext_vector_type(2)))  unsigned  uv2;
typedef __attribute__((ext_vector_type(4)))  float     fv4;
typedef __attribute__((ext_vector_type(4)))  unsigned  u32x4;
typedef __attribute__((ext_vector_type(8)))  int       i32x8;
typedef __attribute__((ext_vector_type(4)))  int       i32x4;

union ABu { v16bf v; uv4 q[2]; unsigned u[8]; };

__device__ __forceinline__ unsigned pkbf(float a, float b) {
  unsigned x = __float_as_uint(a), y = __float_as_uint(b);
  x += 0x7FFFu + ((x >> 16) & 1u);            // round-to-nearest-even-ish
  y += 0x7FFFu + ((y >> 16) & 1u);
  return (x >> 16) | (y & 0xFFFF0000u);
}
__device__ __forceinline__ unsigned short bf16b(float a) {
  unsigned x = __float_as_uint(a);
  x += 0x7FFFu + ((x >> 16) & 1u);
  return (unsigned short)(x >> 16);
}

// ---------- workspace layout (bytes) ----------
static constexpr size_t OFF_OBS = 0;                               // B*T*128 bf16
static constexpr size_t OFF_WC  = (size_t)BB * TT * OBSD * 2;      // [768][128] bf16
static constexpr size_t OFF_WH  = OFF_WC + (size_t)G3 * OBSD * 2;  // [768][256] bf16
static constexpr size_t OFF_WO  = OFF_WH + (size_t)G3 * HID * 2;   // [32][256]  bf16
static constexpr size_t OFF_BC  = OFF_WO + (size_t)32 * HID * 2;   // [768] f32

// ---------- prep kernels ----------
__global__ void k_cvt_obs(const float* __restrict__ in, unsigned short* __restrict__ outp) {
  size_t i = ((size_t)blockIdx.x * blockDim.x + threadIdx.x) * 4;
  fv4 v = *reinterpret_cast<const fv4*>(in + i);
  uv2 p; p.x = pkbf(v.x, v.y); p.y = pkbf(v.z, v.w);
  *reinterpret_cast<uv2*>(outp + i) = p;
}

__global__ void k_wcomb(const float* __restrict__ W_in, const float* __restrict__ b_in,
                        const float* __restrict__ W_ih, const float* __restrict__ b_ih,
                        unsigned short* __restrict__ WcT, float* __restrict__ bcomb) {
  int tid = blockIdx.x * 256 + threadIdx.x;      // 98304 = 768*128
  int n = tid >> 7, k = tid & 127;
  float s = 0.f;
  for (int j = 0; j < PROJ; ++j) s += W_in[k * PROJ + j] * W_ih[j * G3 + n];
  WcT[n * OBSD + k] = bf16b(s);
  if (k == 0) {
    float bc = b_ih[n];
    for (int j = 0; j < PROJ; ++j) bc += b_in[j] * W_ih[j * G3 + n];
    bcomb[n] = bc;
  }
}

__global__ void k_whhT(const float* __restrict__ W_hh, unsigned short* __restrict__ WhT) {
  int n = blockIdx.x, k = threadIdx.x;           // 768 x 256
  WhT[n * HID + k] = bf16b(W_hh[k * G3 + n]);
}

__global__ void k_woutT(const float* __restrict__ W_out, unsigned short* __restrict__ WoT) {
  int n = blockIdx.x, k = threadIdx.x;           // 32 x 256 (pad N 24->32 with zeros)
  WoT[n * HID + k] = (n < LAT) ? bf16b(W_out[k * LAT + n]) : (unsigned short)0;
}

// ---------- main persistent GRU kernel ----------
// grid = 64 blocks (16 batch rows each), 512 threads = 16 waves.
// Wave w owns N-tiles {w, w+16, w+32}: two r/z tiles (fused gx+gh accumulator)
// and one n-gate tile (separate gx / gh accumulators).  4 v8f accs per wave.
// obs A-operand tile for step t+1 is staged into an LDS double buffer by the
// Tensor Data Mover (TENSOR_LOAD_TO_LDS) while step t computes.
#define HS  260    // shH row stride (f32)
#define GS  776    // shG row stride (f32)
#define XS  260    // shX row stride (f32)
#define PS  132    // shHb/shMu row stride (packed bf16 pairs, uint)
#define OS  136    // shObs row stride (bf16): 256B data + 16B TDM pad

// Issue one 2-D TDM tile load: 16 rows x 128 bf16 from gbase (row stride
// T*OBSD elements) into LDS at lds_off, padding +4 DWORDs per 64-DWORD row.
__device__ __forceinline__ void tdm_obs_load(const unsigned short* gbase, unsigned lds_off) {
  unsigned long long ga = (unsigned long long)(size_t)gbase;
  u32x4 g0;
  g0.x = 1u;                                              // count=1 valid descriptor
  g0.y = lds_off;                                         // lds_addr (bytes)
  g0.z = (unsigned)(ga & 0xFFFFFFFFu);                    // global_addr[31:0]
  g0.w = (unsigned)((ga >> 32) & 0x1FFFFFFu) | (2u << 30);// addr[56:32] | type=2
  i32x8 g1;
  g1[0] = (int)((1u << 16) | (1u << 20) | (5u << 22) | (3u << 25));
         // data_size=2B | pad_enable | pad_interval=64dw | pad_amount=4dw
  g1[1] = (int)(128u << 16);   // tensor_dim0 = 128 (low 16)
  g1[2] = (int)(16u  << 16);   // tensor_dim0 hi=0 | tensor_dim1 = 16
  g1[3] = (int)(128u << 16);   // tensor_dim1 hi=0 | tile_dim0 = 128
  g1[4] = 16;                  // tile_dim1 = 16 | tile_dim2 = 0
  g1[5] = TT * OBSD;           // tensor_dim0_stride = 32768 (elements)
  g1[6] = 0;                   // stride hi | tensor_dim1_stride lo
  g1[7] = 0;
  i32x4 z4 = {0, 0, 0, 0};                     // groups 2/3 unused (2-D tensor)
  i32x8 z8 = {0, 0, 0, 0, 0, 0, 0, 0};         // extra group (clang-23 form), unused
  __builtin_amdgcn_tensor_load_to_lds(g0, g1, z4, z4, z8, 0);
}

__global__ __launch_bounds__(512) void k_gru(
    const unsigned short* __restrict__ obsb,   // [B*T][128] bf16
    const unsigned char*  __restrict__ isin,   // [B][T] bool
    const float*          __restrict__ hx,     // [B][256]
    const unsigned short* __restrict__ WcT,    // [768][128] bf16
    const float*          __restrict__ bcomb,  // [768]
    const unsigned short* __restrict__ WhT,    // [768][256] bf16
    const unsigned short* __restrict__ WoT,    // [32][256] bf16
    const float*          __restrict__ bhh,    // [768]
    const float*          __restrict__ bout,   // [24]
    float*                __restrict__ out)    // out[B][T][24] ++ hfinal[B][256]
{
  __shared__ __align__(16) float          shH [16 * HS];   // masked h (f32)
  __shared__ __align__(16) float          shG [16 * GS];   // gx+gh (r,z) | gh (n)
  __shared__ __align__(16) float          shX [16 * XS];   // gx (n-gate)
  __shared__ __align__(16) unsigned       shHb[16 * PS];   // masked h, bf16 pairs
  __shared__ __align__(16) unsigned       shMu[16 * PS];   // mish(h), bf16 pairs
  __shared__ __align__(16) unsigned short shObs[2][16 * OS]; // TDM-staged obs tiles

  const int tid  = threadIdx.x;
  const int w    = tid >> 5;         // 0..15
  const int lane = tid & 31;
  const int n16  = lane & 15;        // col within tile / A row
  const int kh   = lane >> 4;        // lane half
  const int b0   = blockIdx.x * 16;

  // ---- t-invariant hoisting ----
  const int colA = w * 16 + n16;            // r-region column   (0..255)
  const float bA = bcomb[colA] + bhh[colA];
  const float bB = bcomb[colA + 256] + bhh[colA + 256];
  const float bX = bcomb[colA + 512];
  const float bH = bhh[colA + 512];

  const unsigned short* pWc0 = WcT + (size_t)colA * OBSD + kh * 16;
  const unsigned short* pWc1 = pWc0 + (size_t)256 * OBSD;
  const unsigned short* pWc2 = pWc0 + (size_t)512 * OBSD;
  const unsigned short* pWh0 = WhT + (size_t)colA * HID + kh * 16;
  const unsigned short* pWh1 = pWh0 + (size_t)256 * HID;
  const unsigned short* pWh2 = pWh0 + (size_t)512 * HID;

  const unsigned short* obs_row  = obsb + (size_t)(b0 + n16) * TT * OBSD + kh * 8;
  const unsigned short* obs_tile = obsb + (size_t)b0 * TT * OBSD;   // TDM base (t=0)
  const unsigned short* pAh = reinterpret_cast<const unsigned short*>(shHb) + n16 * (2 * PS) + kh * 16;
  const unsigned short* pAm = reinterpret_cast<const unsigned short*>(shMu) + n16 * (2 * PS) + kh * 16;
  const unsigned short* pAo0 = &shObs[0][0] + n16 * OS + kh * 16;
  const unsigned short* pAo1 = &shObs[1][0] + n16 * OS + kh * 16;
  const unsigned lds_obs0 = (unsigned)(size_t)&shObs[0][0];
  const unsigned lds_obs1 = (unsigned)(size_t)&shObs[1][0];

  const int wo = (w < 2) ? w : 0;           // out-GEMM tile (waves 0,1)
  const int colb = wo * 16 + n16;
  const float bo = (colb < LAT) ? bout[colb] : 0.f;
  const unsigned short* pWo = WoT + (size_t)(wo * 16 + n16) * HID + kh * 16;

  const unsigned char* isrow = isin + (size_t)(b0 + (tid & 15)) * TT;
  const int grow = tid & 15;                // gate-phase row
  const int ggrp = tid >> 4;                // gate-phase group 0..31 (8 units each)

  // ---- preload obs tile for t=0 via TDM; init h = hx * (1 - is_init[:,0]) ----
  if (w == 15) tdm_obs_load(obs_tile, lds_obs0);
  {
    float m0 = isrow[0] ? 0.f : 1.f;
#pragma unroll
    for (int p = 0; p < 4; ++p) {
      int u = ggrp * 8 + 2 * p;
      float h0 = hx[(size_t)(b0 + grow) * HID + u] * m0;
      float h1 = hx[(size_t)(b0 + grow) * HID + u + 1] * m0;
      shH[grow * HS + u] = h0;
      shH[grow * HS + u + 1] = h1;
      shHb[grow * PS + (u >> 1)] = pkbf(h0, h1);
    }
  }
  if (w == 15) __builtin_amdgcn_s_wait_tensorcnt(0);
  __syncthreads();

  auto do_out = [&](int tp) {   // y[:, tp, :] = mish(h_tp) @ W_out + b_out
    v8f acc;
#pragma unroll
    for (int r = 0; r < 8; ++r) acc[r] = bo;
#pragma unroll 1
    for (int kk = 0; kk < 8; ++kk) {
      ABu A, Bf;
      A.q[0] = *reinterpret_cast<const uv4*>(pAm + kk * 32);
      A.q[1] = *reinterpret_cast<const uv4*>(pAm + kk * 32 + 8);
      Bf.q[0] = *reinterpret_cast<const uv4*>(pWo + kk * 32);
      Bf.q[1] = *reinterpret_cast<const uv4*>(pWo + kk * 32 + 8);
      acc = __builtin_amdgcn_wmma_f32_16x16x32_bf16(false, A.v, false, Bf.v,
                                                    (short)0, acc, false, false);
    }
    if (colb < LAT) {
#pragma unroll
      for (int r = 0; r < 8; ++r)
        out[((size_t)(b0 + r + 8 * kh) * TT + tp) * LAT + colb] = acc[r];
    }
  };

  for (int t = 0; t < TT; ++t) {
    // ===== GEMM phase =====
    // Kick the Tensor Data Mover for step t+1's obs tile (double buffered).
    if (w == 15 && t + 1 < TT)
      tdm_obs_load(obs_tile + (size_t)(t + 1) * OBSD, ((t + 1) & 1) ? lds_obs1 : lds_obs0);

    if (w < 2 && t > 0) do_out(t - 1);

    v8f aA, aB, aX, aH;     // r-tile, z-tile, gx(n)-tile, gh(n)-tile
#pragma unroll
    for (int r = 0; r < 8; ++r) { aA[r] = bA; aB[r] = bB; aX[r] = bX; aH[r] = bH; }

    if (t + 2 < TT)  // warm GL2 one tile ahead of the TDM
      __builtin_prefetch(obs_row + (size_t)(t + 2) * OBSD, 0, 3);

    // gx_t = obs_t @ W_comb  (K=128, A = TDM-staged bf16 obs tile in LDS)
    const unsigned short* pa = (t & 1) ? pAo1 : pAo0;
#pragma unroll 2
    for (int kk = 0; kk < 4; ++kk) {
      ABu A, B0, B1, B2;
      A.q[0] = *reinterpret_cast<const uv4*>(pa + kk * 32);
      A.q[1] = *reinterpret_cast<const uv4*>(pa + kk * 32 + 8);
      B0.q[0] = *reinterpret_cast<const uv4*>(pWc0 + kk * 32);
      B0.q[1] = *reinterpret_cast<const uv4*>(pWc0 + kk * 32 + 8);
      B1.q[0] = *reinterpret_cast<const uv4*>(pWc1 + kk * 32);
      B1.q[1] = *reinterpret_cast<const uv4*>(pWc1 + kk * 32 + 8);
      B2.q[0] = *reinterpret_cast<const uv4*>(pWc2 + kk * 32);
      B2.q[1] = *reinterpret_cast<const uv4*>(pWc2 + kk * 32 + 8);
      aA = __builtin_amdgcn_wmma_f32_16x16x32_bf16(false, A.v, false, B0.v, (short)0, aA, false, false);
      aB = __builtin_amdgcn_wmma_f32_16x16x32_bf16(false, A.v, false, B1.v, (short)0, aB, false, false);
      aX = __builtin_amdgcn_wmma_f32_16x16x32_bf16(false, A.v, false, B2.v, (short)0, aX, false, false);
    }

    // gh_t = h @ W_hh  (K=256, A = packed bf16 h from LDS)
#pragma unroll 2
    for (int kk = 0; kk < 8; ++kk) {
      ABu A, B0, B1, B2;
      A.q[0] = *reinterpret_cast<const uv4*>(pAh + kk * 32);
      A.q[1] = *reinterpret_cast<const uv4*>(pAh + kk * 32 + 8);
      B0.q[0] = *reinterpret_cast<const uv4*>(pWh0 + kk * 32);
      B0.q[1] = *reinterpret_cast<const uv4*>(pWh0 + kk * 32 + 8);
      B1.q[0] = *reinterpret_cast<const uv4*>(pWh1 + kk * 32);
      B1.q[1] = *reinterpret_cast<const uv4*>(pWh1 + kk * 32 + 8);
      B2.q[0] = *reinterpret_cast<const uv4*>(pWh2 + kk * 32);
      B2.q[1] = *reinterpret_cast<const uv4*>(pWh2 + kk * 32 + 8);
      aA = __builtin_amdgcn_wmma_f32_16x16x32_bf16(false, A.v, false, B0.v, (short)0, aA, false, false);
      aB = __builtin_amdgcn_wmma_f32_16x16x32_bf16(false, A.v, false, B1.v, (short)0, aB, false, false);
      aH = __builtin_amdgcn_wmma_f32_16x16x32_bf16(false, A.v, false, B2.v, (short)0, aH, false, false);
    }

    // spill accumulators to LDS (D layout: row = r + 8*kh, col = tile*16 + n16)
#pragma unroll
    for (int r = 0; r < 8; ++r) {
      int row = r + 8 * kh;
      shG[row * GS + colA]       = aA[r];
      shG[row * GS + colA + 256] = aB[r];
      shG[row * GS + colA + 512] = aH[r];
      shX[row * XS + colA]       = aX[r];
    }
    __syncthreads();

    // ===== gate phase (f32 VALU) =====
    {
      bool rn = (t + 1 < TT) && isrow[t + 1];
      const float* gRow = &shG[grow * GS];
      const float* xRow = &shX[grow * XS];
      float* hRow = &shH[grow * HS];
#pragma unroll
      for (int p = 0; p < 4; ++p) {
        int u = ggrp * 8 + 2 * p;
        float hn2[2], mi2[2];
#pragma unroll
        for (int e = 0; e < 2; ++e) {
          int ue = u + e;
          float r  = 1.f / (1.f + expf(-gRow[ue]));
          float z  = 1.f / (1.f + expf(-gRow[256 + ue]));
          float nn = tanhf(xRow[ue] + r * gRow[512 + ue]);
          float hn = (1.f - z) * nn + z * hRow[ue];
          float sp = fmaxf(hn, 0.f) + log1pf(expf(-fabsf(hn)));   // softplus
          mi2[e] = hn * tanhf(sp);                                // mish(h)
          hn2[e] = rn ? 0.f : hn;                                 // pre-mask for t+1
        }
        hRow[u]     = hn2[0];
        hRow[u + 1] = hn2[1];
        shHb[grow * PS + (u >> 1)] = pkbf(hn2[0], hn2[1]);
        shMu[grow * PS + (u >> 1)] = pkbf(mi2[0], mi2[1]);
      }
    }
    // TDM tile for t+1 must be visible to all waves after this barrier.
    if (w == 15 && t + 1 < TT) __builtin_amdgcn_s_wait_tensorcnt(0);
    __syncthreads();
  }

  // tail: y for t = T-1, then h_final (unmasked last h)
  if (w < 2) do_out(TT - 1);
  {
#pragma unroll
    for (int p = 0; p < 8; ++p) {
      int u = ggrp * 8 + p;
      out[(size_t)BB * TT * LAT + (size_t)(b0 + grow) * HID + u] = shH[grow * HS + u];
    }
  }
}

// ---------- launcher ----------
extern "C" void kernel_launch(void* const* d_in, const int* in_sizes, int n_in,
                              void* d_out, int out_size, void* d_ws, size_t ws_size,
                              hipStream_t stream) {
  (void)in_sizes; (void)n_in; (void)out_size; (void)ws_size;
  const float*         obs    = (const float*)d_in[0];
  const unsigned char* is_init= (const unsigned char*)d_in[1];
  const float*         hx     = (const float*)d_in[2];
  const float*         W_in   = (const float*)d_in[3];
  const float*         b_in   = (const float*)d_in[4];
  const float*         W_ih   = (const float*)d_in[5];
  const float*         b_ih   = (const float*)d_in[6];
  const float*         W_hh   = (const float*)d_in[7];
  const float*         b_hh   = (const float*)d_in[8];
  const float*         W_out  = (const float*)d_in[9];
  const float*         b_out  = (const float*)d_in[10];
  float* out = (float*)d_out;

  char* ws = (char*)d_ws;
  unsigned short* obsb = (unsigned short*)(ws + OFF_OBS);
  unsigned short* WcT  = (unsigned short*)(ws + OFF_WC);
  unsigned short* WhT  = (unsigned short*)(ws + OFF_WH);
  unsigned short* WoT  = (unsigned short*)(ws + OFF_WO);
  float*          bcmb = (float*)(ws + OFF_BC);

  k_cvt_obs<<<(BB * TT * OBSD) / (256 * 4), 256, 0, stream>>>(obs, obsb);
  k_wcomb<<<(G3 * OBSD) / 256, 256, 0, stream>>>(W_in, b_in, W_ih, b_ih, WcT, bcmb);
  k_whhT<<<G3, HID, 0, stream>>>(W_hh, WhT);
  k_woutT<<<32, HID, 0, stream>>>(W_out, WoT);
  k_gru<<<BB / 16, 512, 0, stream>>>(obsb, is_init, hx, WcT, bcmb, WhT, WoT,
                                     b_hh, b_out, out);
}
